// GCN_76201309766160
// MI455X (gfx1250) — matrix operand order repaired and checked by reading, not compiled, not run on hardware
//
#include <hip/hip_runtime.h>
#include <stdint.h>

typedef __attribute__((ext_vector_type(16))) __bf16 v16bf;
typedef __attribute__((ext_vector_type(8)))  __bf16 v8bf;
typedef __attribute__((ext_vector_type(8)))  float  v8f;

#define FEAT 128
#define LDS_STRIDE 136   // bf16 elems per row: 272B -> 16B aligned chunks, bank-spread

// ---------------- utility kernels ----------------

__global__ void k_zero(float* __restrict__ p, long n) {
  long i = (long)blockIdx.x * blockDim.x + threadIdx.x;
  if (i < n) p[i] = 0.0f;
}

__global__ void k_deg(const int* __restrict__ src, const int* __restrict__ dst,
                      float* __restrict__ degO, float* __restrict__ degI, int E) {
  int e = blockIdx.x * blockDim.x + threadIdx.x;
  if (e < E) {
    atomicAdd(&degO[src[e]], 1.0f);
    atomicAdd(&degI[dst[e]], 1.0f);
  }
}

__global__ void k_norm(float* __restrict__ ns, float* __restrict__ nd, int n) {
  int i = blockIdx.x * blockDim.x + threadIdx.x;
  if (i < n) {
    ns[i] = rsqrtf(fmaxf(ns[i], 1.0f));
    nd[i] = rsqrtf(fmaxf(nd[i], 1.0f));
  }
}

// Pack W (K x N, f32 row-major) -> Wt (N x K, bf16 row-major).
// Each WMMA B-fragment lane then reads one contiguous 32B chunk of a column.
__global__ void k_packW(const float* __restrict__ W, __bf16* __restrict__ Wt) {
  int i = blockIdx.x * blockDim.x + threadIdx.x;   // 0 .. 16383
  int n = i >> 7, k = i & 127;
  Wt[n * FEAT + k] = (__bf16)W[k * FEAT + n];
}

// Scatter: agg[dst] += x[src] * ns[src].  32 lanes per edge, float4 per lane.
__global__ void k_scatter(const float* __restrict__ x, const int* __restrict__ src,
                          const int* __restrict__ dst, const float* __restrict__ ns,
                          float* __restrict__ agg, int E) {
  long t = (long)blockIdx.x * blockDim.x + threadIdx.x;
  long e = t >> 5;
  if (e >= E) return;
  int c = ((int)t & 31) << 2;
  int s = src[e], d = dst[e];
  float sc = ns[s];
  const float4 v = *(const float4*)(x + (long)s * FEAT + c);
  float* o = agg + (long)d * FEAT + c;
  atomicAdd(o + 0, v.x * sc);
  atomicAdd(o + 1, v.y * sc);
  atomicAdd(o + 2, v.z * sc);
  atomicAdd(o + 3, v.w * sc);
}

// ---------------- WMMA GEMM: h = relu((agg * nd) @ W + b), in place ----------------
// Block = 256 threads = 8 waves. Block owns 16 rows; wave w owns columns [16w,16w+16).
// K = 128 consumed as 4 chained v_wmma_f32_16x16x32_bf16.
__global__ void __launch_bounds__(256)
k_gemm(float* __restrict__ agg, const float* __restrict__ nd,
       const __bf16* __restrict__ Wt, const float* __restrict__ bias) {
  __shared__ __bf16 xs[16 * LDS_STRIDE];

  const int r0  = blockIdx.x * 16;
  const int tid = threadIdx.x;

  // Stage 16x128 activation tile: scale by dest norm, convert to bf16.
#pragma unroll
  for (int i = 0; i < 8; ++i) {
    int idx = tid + i * 256;          // 0..2047
    int r = idx >> 7, c = idx & 127;
    float v = agg[(long)(r0 + r) * FEAT + c] * nd[r0 + r];
    xs[r * LDS_STRIDE + c] = (__bf16)v;
  }
  __syncthreads();

  const int wave = tid >> 5;          // 0..7 -> column tile
  const int lane = tid & 31;
  const int hi   = lane >> 4;         // 0 | 1 (half-wave)
  const int mrow = lane & 15;         // A row / C column index
  const int ncol = wave * 16 + mrow;  // output column for B & C

  v8f acc;
  const float bv = bias[ncol];
#pragma unroll
  for (int i = 0; i < 8; ++i) acc[i] = bv;

  const __bf16* wrow = Wt + (long)ncol * FEAT;

#pragma unroll
  for (int kt = 0; kt < 4; ++kt) {
    // A fragment (contiguous k order used here, same as B): lane<16 -> K 0..15, lane>=16 -> K 16..31
    const int ka = kt * 32 + hi * 16;
    const __bf16* ap = &xs[mrow * LDS_STRIDE + ka];
    v8bf alo = *(const v8bf*)ap;
    v8bf ahi = *(const v8bf*)(ap + 8);
    v16bf a;
#pragma unroll
    for (int i = 0; i < 8; ++i) { a[i] = alo[i]; a[8 + i] = ahi[i]; }

    // B fragment (contiguous k order used here, same as A): lane<16 -> K 0..15 of column ncol, lane>=16 -> K 16..31
    const int kb = kt * 32 + hi * 16;
    v16bf b = *(const v16bf*)(wrow + kb);   // 32B aligned contiguous chunk

    acc = __builtin_amdgcn_wmma_f32_16x16x32_bf16(
        /*neg_a=*/false, a, /*neg_b=*/false, b,
        /*c_mod=*/(short)0, acc, /*reuse_a=*/false, /*reuse_b=*/false);
  }

  // C/D layout: VGPR i -> M = i (lanes 0-15) or i+8 (lanes 16-31); N = lane&15.
#pragma unroll
  for (int i = 0; i < 8; ++i) {
    int m = i + hi * 8;
    agg[(long)(r0 + m) * FEAT + ncol] = fmaxf(acc[i], 0.0f);
  }
}

__global__ void k_out(const float* __restrict__ h, float* __restrict__ out, long n) {
  long i = (long)blockIdx.x * blockDim.x + threadIdx.x;
  if (i < n) {
    float v = h[i];
    out[i]     = v;
    out[n + i] = (v >= 0.5f) ? 1.0f : 0.0f;
  }
}

// ---------------- launcher ----------------

extern "C" void kernel_launch(void* const* d_in, const int* in_sizes, int n_in,
                              void* d_out, int out_size, void* d_ws, size_t ws_size,
                              hipStream_t stream) {
  const float* in_feat = (const float*)d_in[0];
  const int*   src     = (const int*)d_in[1];
  const int*   dst     = (const int*)d_in[2];
  const float* Ws[5] = {(const float*)d_in[3], (const float*)d_in[5], (const float*)d_in[7],
                        (const float*)d_in[9], (const float*)d_in[11]};
  const float* bs[5] = {(const float*)d_in[4], (const float*)d_in[6], (const float*)d_in[8],
                        (const float*)d_in[10], (const float*)d_in[12]};

  const int  Nn = in_sizes[0] / FEAT;       // 50000
  const int  E  = in_sizes[1];              // 800000
  const long NF = (long)Nn * FEAT;

  float* bufA = (float*)d_ws;
  float* bufB = bufA + NF;
  float* ns   = bufB + NF;
  float* nd   = ns + Nn;
  uintptr_t wt_addr = ((uintptr_t)(nd + Nn) + 63) & ~(uintptr_t)63;
  __bf16* Wt = (__bf16*)wt_addr;            // 128*128 bf16, 64B aligned

  // degrees -> symmetric norms
  k_zero<<<(int)((2L * Nn + 255) / 256), 256, 0, stream>>>(ns, 2L * Nn);
  k_deg <<<(E + 255) / 256, 256, 0, stream>>>(src, dst, ns, nd, E);
  k_norm<<<(Nn + 255) / 256, 256, 0, stream>>>(ns, nd, Nn);

  const float* x = in_feat;
  float* bufs[2] = {bufA, bufB};
  const long scatter_threads = (long)E * 32;

  for (int l = 0; l < 5; ++l) {
    float* agg = bufs[l & 1];               // l0->A, l1->B, l2->A, ... (l4 ends in A)
    k_packW<<<64, 256, 0, stream>>>(Ws[l], Wt);
    k_zero <<<(int)((NF + 255) / 256), 256, 0, stream>>>(agg, NF);
    k_scatter<<<(int)((scatter_threads + 255) / 256), 256, 0, stream>>>(x, src, dst, ns, agg, E);
    k_gemm<<<Nn / 16, 256, 0, stream>>>(agg, nd, Wt, bs[l]);
    x = agg;
  }

  k_out<<<(int)((NF + 255) / 256), 256, 0, stream>>>(bufA, (float*)d_out, NF);
}
